// retrieval_module_43671227465961
// MI455X (gfx1250) — matrix operand order repaired and checked by reading, not compiled, not run on hardware
//
#include <hip/hip_runtime.h>
#include <math.h>

// ---------------- types for CDNA5 WMMA ----------------
typedef __attribute__((ext_vector_type(16))) __bf16 v16bf;
typedef __attribute__((ext_vector_type(8)))  __bf16 v8bf;
typedef __attribute__((ext_vector_type(4)))  __bf16 v4bf;
typedef __attribute__((ext_vector_type(8)))  float  v8f;

#define PDIM 256
#define B_IMG 512
#define N_SPOT 200000
#define DIN_IMG 1024
#define DIN_SPOT 300
#define K_TOP 50
#define NUM_TYPE 10
#define TYPE_VOCAB 30
#define CHUNK 896                  // sim columns per block (16x896x4B = 56KB LDS)
#define NCH 224                    // 224*896 = 200704 >= 200000
#define CAND (NCH * K_TOP)         // 11200 candidates per image row
#define NEGV (-1e30f)

static __device__ __forceinline__ v8f wmma_bf16(v16bf a, v16bf b, v8f c) {
    // D = A(16x32 bf16) * B(32x16 bf16) + C(16x16 f32)
    return __builtin_amdgcn_wmma_f32_16x16x32_bf16(false, a, false, b,
                                                   (short)0, c, false, false);
}

// two contiguous 8x bf16 (16B) segments -> one 16-element fragment
static __device__ __forceinline__ v16bf ld2x8(const __bf16* p0, const __bf16* p1) {
    v8bf lo = *(const v8bf*)p0;
    v8bf hi = *(const v8bf*)p1;
    return __builtin_shufflevector(lo, hi,
        0,1,2,3,4,5,6,7,8,9,10,11,12,13,14,15);
}

// =====================================================================
// Projection head: z = xW1+b1; h = gelu(z); h = hW2+b2+z; LN; L2-norm
// One block = 128 threads (4 waves) computes 16 rows x 256 cols.
// LDS layouts arranged so all WMMA fragment reads are ds_load_b128.
// =====================================================================
template<int DIN>
__global__ __launch_bounds__(128)
void proj_norm_kernel(const float* __restrict__ x,
                      const float* __restrict__ w1, const float* __restrict__ b1,
                      const float* __restrict__ w2, const float* __restrict__ b2,
                      const float* __restrict__ gam, const float* __restrict__ bet,
                      __bf16* __restrict__ outN)
{
    __shared__ alignas(16) __bf16 aT[16][32];      // A tile, rows 64B
    __shared__ alignas(16) __bf16 wTt[256][40];    // W tile TRANSPOSED [n][k], row 80B
    __shared__ alignas(16) __bf16 hT[16][264];     // h (bf16), row 528B = 33*16
    __shared__ float  hv[16][264];
    __shared__ float  s_mu[16], s_rstd[16], s_rinv[16];

    const int tid  = threadIdx.x;
    const int lane = tid & 31;
    const int wv   = tid >> 5;        // 0..3, each wave owns a 64-col strip
    const int half = lane >> 4;
    const int lr   = lane & 15;
    const int rowbase = blockIdx.x * 16;

    v8f accZ[4], accH[4];
#pragma unroll
    for (int t = 0; t < 4; ++t) accZ[t] = (v8f)0.f;

    // ---------------- GEMM 1: z = x @ w1 ----------------
    const int nK = (DIN + 31) / 32;
    for (int kt = 0; kt < nK; ++kt) {
        const int k0 = kt * 32;
#pragma unroll
        for (int i = 0; i < 4; ++i) {               // A tile 16x32 -> LDS (bf16)
            int e = tid + 128 * i;
            int r = e >> 5, c = e & 31, gk = k0 + c;
            float v = (gk < DIN) ? x[(size_t)(rowbase + r) * DIN + gk] : 0.f;
            aT[r][c] = (__bf16)v;
        }
#pragma unroll
        for (int i = 0; i < 16; ++i) {              // W1 tile -> LDS transposed
            int e = tid + 128 * i;
            int n = e & 255, r = (e >> 8) * 4;      // 4 consecutive k per thread
            v4bf w4;
#pragma unroll
            for (int ii = 0; ii < 4; ++ii) {
                int gk = k0 + r + ii;
                float v = (gk < DIN) ? w1[(size_t)gk * PDIM + n] : 0.f;
                w4[ii] = (__bf16)v;
            }
            *(v4bf*)&wTt[n][r] = w4;                // 8B LDS store
        }
        __syncthreads();
        // A frag: 16-bit A 16x32 layout -> two 16B LDS loads
        v16bf a = ld2x8(&aT[lr][half * 8], &aT[lr][16 + half * 8]);
#pragma unroll
        for (int t = 0; t < 4; ++t) {
            int n = wv * 64 + t * 16 + lr;
            // B frag: K = half*8 + j, then 16 + half*8 + j (matches A order)
            v16bf b = ld2x8(&wTt[n][half * 8], &wTt[n][16 + half * 8]);
            accZ[t] = wmma_bf16(a, b, accZ[t]);
        }
        __syncthreads();
    }

    // bias + exact-erf GELU; keep z for residual; stash h as bf16 in LDS
    float zf[4][8];
#pragma unroll
    for (int t = 0; t < 4; ++t) {
        int n = wv * 64 + t * 16 + lr;
        float bb = b1[n];
#pragma unroll
        for (int r = 0; r < 8; ++r) {
            float z = accZ[t][r] + bb;
            zf[t][r] = z;
            float h = 0.5f * z * (1.f + erff(z * 0.70710678118654752f));
            hT[r + half * 8][n] = (__bf16)h;        // C layout: M = r + 8*half
        }
    }
    __syncthreads();

    // ---------------- GEMM 2: h @ w2 (K = 256) ----------------
#pragma unroll
    for (int t = 0; t < 4; ++t) accH[t] = (v8f)0.f;
    for (int kt = 0; kt < 8; ++kt) {
        const int k0 = kt * 32;
#pragma unroll
        for (int i = 0; i < 16; ++i) {              // W2 tile -> LDS transposed
            int e = tid + 128 * i;
            int n = e & 255, r = (e >> 8) * 4;
            v4bf w4;
#pragma unroll
            for (int ii = 0; ii < 4; ++ii)
                w4[ii] = (__bf16)w2[(size_t)(k0 + r + ii) * PDIM + n];
            *(v4bf*)&wTt[n][r] = w4;
        }
        __syncthreads();
        v16bf a = ld2x8(&hT[lr][k0 + half * 8], &hT[lr][k0 + 16 + half * 8]);
#pragma unroll
        for (int t = 0; t < 4; ++t) {
            int n = wv * 64 + t * 16 + lr;
            v16bf b = ld2x8(&wTt[n][half * 8], &wTt[n][16 + half * 8]);
            accH[t] = wmma_bf16(a, b, accH[t]);
        }
        __syncthreads();
    }

    // residual + bias -> hv (f32)
#pragma unroll
    for (int t = 0; t < 4; ++t) {
        int n = wv * 64 + t * 16 + lr;
        float bb = b2[n];
#pragma unroll
        for (int r = 0; r < 8; ++r)
            hv[r + half * 8][n] = accH[t][r] + bb + zf[t][r];
    }
    __syncthreads();

    // LayerNorm stats + L2 norm of LN output (one thread per row)
    if (tid < 16) {
        float s = 0.f, sq = 0.f;
        for (int c = 0; c < PDIM; ++c) { float v = hv[tid][c]; s += v; sq += v * v; }
        float mu  = s / PDIM;
        float var = sq / PDIM - mu * mu;            // population variance (jnp.var)
        float rstd = rsqrtf(var + 1e-5f);
        float l2 = 0.f;
        for (int c = 0; c < PDIM; ++c) {
            float ln = gam[c] * (hv[tid][c] - mu) * rstd + bet[c];
            l2 += ln * ln;
        }
        s_mu[tid] = mu; s_rstd[tid] = rstd;
        s_rinv[tid] = 1.f / fmaxf(sqrtf(l2), 1e-12f);
    }
    __syncthreads();

    for (int i = 0; i < 32; ++i) {                  // normalized output, bf16
        int e = tid + 128 * i;
        int r = e >> 8, c = e & 255;
        float ln = gam[c] * (hv[r][c] - s_mu[r]) * s_rstd[r] + bet[c];
        outN[(size_t)(rowbase + r) * PDIM + c] = (__bf16)(ln * s_rinv[r]);
    }
}

// =====================================================================
// Similarity GEMM + per-chunk top-50.
// Block (256 threads): 16 image rows x CHUNK spots.  grid = (NCH, 32).
// =====================================================================
__global__ __launch_bounds__(256)
void sim_topk_kernel(const __bf16* __restrict__ imgN, const __bf16* __restrict__ exprN,
                     float* __restrict__ candV, int* __restrict__ candI)
{
    __shared__ float simb[16][CHUNK];               // 56 KB

    const int tid  = threadIdx.x;
    const int lane = tid & 31;
    const int wv   = tid >> 5;                      // 0..7
    const int half = lane >> 4;
    const int lr   = lane & 15;
    const int chunk   = blockIdx.x;
    const int rowbase = blockIdx.y * 16;

    // A fragments for this 16-row block (K = 256 -> 8 frags), direct 16B loads
    v16bf afr[8];
    {
        const __bf16* arow = imgN + (size_t)(rowbase + lr) * PDIM;
#pragma unroll
        for (int kt = 0; kt < 8; ++kt) {
            const __bf16* p0 = arow + kt * 32 + half * 8;
            afr[kt] = ld2x8(p0, p0 + 16);
        }
    }

    // 56 column tiles, 7 per wave
    for (int t = 0; t < 7; ++t) {
        const int tau  = wv * 7 + t;
        const int colb = tau * 16;
        int spot  = chunk * CHUNK + colb + lr;
        int spotc = spot < N_SPOT ? spot : (N_SPOT - 1);
        const __bf16* brow = exprN + (size_t)spotc * PDIM;
        v8f acc = (v8f)0.f;
#pragma unroll
        for (int kt = 0; kt < 8; ++kt) {
            const __bf16* p0 = brow + kt * 32 + half * 8;
            v16bf b = ld2x8(p0, p0 + 16);
            acc = wmma_bf16(afr[kt], b, acc);
        }
        bool valid = spot < N_SPOT;
#pragma unroll
        for (int r = 0; r < 8; ++r)
            simb[r + half * 8][colb + lr] = valid ? acc[r] : NEGV;
    }
    __syncthreads();

    // per-wave iterative argmax: 2 rows per wave, top-50 per row
    for (int rr = 0; rr < 2; ++rr) {
        const int row  = wv * 2 + rr;
        const int grow = rowbase + row;
        for (int k = 0; k < K_TOP; ++k) {
            float bv = NEGV; int bi = 0x7fffffff;
            for (int i = lane; i < CHUNK; i += 32) {
                float v = simb[row][i];
                if (v > bv || (v == bv && i < bi)) { bv = v; bi = i; }
            }
#pragma unroll
            for (int off = 16; off; off >>= 1) {     // wave32 butterfly
                float ov = __shfl_xor(bv, off, 32);
                int   oi = __shfl_xor(bi, off, 32);
                if (ov > bv || (ov == bv && oi < bi)) { bv = ov; bi = oi; }
            }
            if (lane == 0) {
                simb[row][bi] = NEGV;
                size_t o = ((size_t)grow * NCH + chunk) * K_TOP + k;
                candV[o] = bv;
                candI[o] = chunk * CHUNK + bi;
            }
        }
    }
}

// =====================================================================
// Per image row: merge candidates -> global top-50, conf mask, type
// histogram top-10 filter, weighted mean over gathered all_expr rows.
// =====================================================================
__global__ __launch_bounds__(256)
void merge_kernel(const float* __restrict__ candV, const int* __restrict__ candI,
                  const int* __restrict__ cellType, const float* __restrict__ allExpr,
                  float* __restrict__ out)
{
    __shared__ float mv[CAND];                      // 44.8 KB
    __shared__ float rv[256];
    __shared__ int   ri[256];
    __shared__ float topv[K_TOP];
    __shared__ int   topi[K_TOP];
    __shared__ int   tsel[K_TOP];
    __shared__ int   cnts[TYPE_VOCAB];
    __shared__ int   ttyp[NUM_TYPE];
    __shared__ int   tcnt[NUM_TYPE];
    __shared__ int   keep[K_TOP];

    const int tid = threadIdx.x;
    const int row = blockIdx.x;
    const size_t cbase = (size_t)row * CAND;

    for (int i = tid; i < CAND; i += 256) mv[i] = candV[cbase + i];
    __syncthreads();

    // global top-50 (position order == ascending spot index for tie-break)
    for (int k = 0; k < K_TOP; ++k) {
        float bv = NEGV; int bi = 0x7fffffff;
        for (int i = tid; i < CAND; i += 256) {
            float v = mv[i];
            if (v > bv || (v == bv && i < bi)) { bv = v; bi = i; }
        }
        rv[tid] = bv; ri[tid] = bi;
        __syncthreads();
        for (int s = 128; s; s >>= 1) {
            if (tid < s) {
                float ov = rv[tid + s]; int oi = ri[tid + s];
                if (ov > rv[tid] || (ov == rv[tid] && oi < ri[tid])) {
                    rv[tid] = ov; ri[tid] = oi;
                }
            }
            __syncthreads();
        }
        if (tid == 0) {
            int p = ri[0];
            topv[k] = rv[0];
            topi[k] = candI[cbase + p];
            mv[p] = NEGV;
        }
        __syncthreads();
    }

    if (tid < TYPE_VOCAB) cnts[tid] = 0;
    __syncthreads();
    if (tid < K_TOP) {
        int t = cellType[topi[tid]];
        tsel[tid] = t;
        atomicAdd(&cnts[t], 1);
    }
    if (tid == 0) {                                 // conf / mask
        float s = 0.f;
        for (int k = 0; k < K_TOP; ++k) s += topv[k];
        out[row] = ((s / K_TOP) > 0.1f) ? 1.f : 0.f;
    }
    __syncthreads();
    if (tid == 0) {                                 // top-10 types, low-id tie-break
        for (int j = 0; j < NUM_TYPE; ++j) {
            int bc = -1, bt = 0;
            for (int t = 0; t < TYPE_VOCAB; ++t)
                if (cnts[t] > bc) { bc = cnts[t]; bt = t; }
            ttyp[j] = bt; tcnt[j] = bc; cnts[bt] = -1;
        }
    }
    __syncthreads();
    if (tid < K_TOP) {
        int kp = 0;
        for (int j = 0; j < NUM_TYPE; ++j)
            if (ttyp[j] == tsel[tid] && tcnt[j] > 0) kp = 1;
        keep[tid] = kp;
    }
    __syncthreads();

    float wsum = 0.f;
    for (int k = 0; k < K_TOP; ++k) wsum += (float)keep[k];
    for (int d = tid; d < DIN_SPOT; d += 256) {
        float s = 0.f;
        for (int k = 0; k < K_TOP; ++k)
            if (keep[k]) s += allExpr[(size_t)topi[k] * DIN_SPOT + d];
        out[B_IMG + (size_t)row * DIN_SPOT + d] = s / wsum;
    }
}

// =====================================================================
extern "C" void kernel_launch(void* const* d_in, const int* in_sizes, int n_in,
                              void* d_out, int out_size, void* d_ws, size_t ws_size,
                              hipStream_t stream) {
    const float* feature  = (const float*)d_in[0];
    const float* all_expr = (const float*)d_in[1];
    const int*   all_ct   = (const int*)d_in[2];
    const float* iw1 = (const float*)d_in[3];
    const float* ib1 = (const float*)d_in[4];
    const float* iw2 = (const float*)d_in[5];
    const float* ib2 = (const float*)d_in[6];
    const float* ig  = (const float*)d_in[7];
    const float* ibt = (const float*)d_in[8];
    const float* sw1 = (const float*)d_in[9];
    const float* sb1 = (const float*)d_in[10];
    const float* sw2 = (const float*)d_in[11];
    const float* sb2 = (const float*)d_in[12];
    const float* sg  = (const float*)d_in[13];
    const float* sbt = (const float*)d_in[14];
    float* out = (float*)d_out;

    // workspace layout (needs ~148.5 MB):
    //   imgN  bf16 [512,256]      @ 0          (262144 B)
    //   exprN bf16 [200000,256]   @ 262144     (102400000 B)
    //   candV f32  [512*11200]    @ +          (22937600 B)
    //   candI i32  [512*11200]    @ +          (22937600 B)
    char* ws = (char*)d_ws;
    __bf16* imgN  = (__bf16*)ws;
    __bf16* exprN = (__bf16*)(ws + 262144);
    float*  candV = (float*)(ws + 262144 + 102400000ull);
    int*    candI = (int*)(ws + 262144 + 102400000ull + 22937600ull);

    proj_norm_kernel<DIN_IMG><<<B_IMG / 16, 128, 0, stream>>>(
        feature, iw1, ib1, iw2, ib2, ig, ibt, imgN);
    proj_norm_kernel<DIN_SPOT><<<N_SPOT / 16, 128, 0, stream>>>(
        all_expr, sw1, sb1, sw2, sb2, sg, sbt, exprN);

    dim3 g(NCH, B_IMG / 16);
    sim_topk_kernel<<<g, 256, 0, stream>>>(imgN, exprN, candV, candI);
    merge_kernel<<<B_IMG, 256, 0, stream>>>(candV, candI, all_ct, all_expr, out);
}
